// KeySelect_81363860455653
// MI455X (gfx1250) — matrix operand order repaired and checked by reading, not compiled, not run on hardware
//
#include <hip/hip_runtime.h>
#include <hip/hip_bf16.h>

typedef __attribute__((ext_vector_type(16))) _Float16 v16h;
typedef __attribute__((ext_vector_type(8)))  float    v8f;

template <typename T> struct Vec4;
template <> struct Vec4<float>    { using type = float    __attribute__((ext_vector_type(4))); };
template <> struct Vec4<_Float16> { using type = _Float16 __attribute__((ext_vector_type(4))); };

// ---------------------------------------------------------------------------
// Pack conv weights: fold BN scale, convert to f16, keep each 32-ci group in ci order
// so each WMMA lane reads its 16 A-halves contiguously.
// A-layout used (16-bit A 16x32): lanes 0-15 take k 0..15, lanes 16-31 take
// k 16..31.  Storage position p(r) = r.
// Destination layout [cout][cc][tap][32]: all 9 taps of a (cout, chunk) pair
// are contiguous (64 B apart) so the conv kernel addresses them with one base
// pointer + immediate offsets.
// Also emits per-cout bias = beta - mean*scale.
// ---------------------------------------------------------------------------
__global__ __launch_bounds__(256) void pack_weights_kernel(
    const float* __restrict__ w, const float* __restrict__ bn,
    _Float16* __restrict__ wp, float* __restrict__ bias, int Cout, int Cin) {
  int idx = blockIdx.x * 256 + threadIdx.x;
  if (idx < Cout) {
    float g = bn[idx], be = bn[Cout + idx], m = bn[2 * Cout + idx], v = bn[3 * Cout + idx];
    float s = g * rsqrtf(v + 1e-5f);
    bias[idx] = be - m * s;
  }
  int total = 9 * Cout * Cin;
  if (idx >= total) return;
  int ci = idx % Cin;
  int rest = idx / Cin;
  int co = rest % Cout;
  int tap = rest / Cout;                // tap = kh*3 + kw
  float g = bn[co], v = bn[3 * Cout + co];
  float s = g * rsqrtf(v + 1e-5f);
  int kh = tap / 3, kw = tap % 3;
  float val = w[(((size_t)co * Cin + ci) * 3 + kh) * 3 + kw] * s;
  int cc = ci >> 5, r = ci & 31;
  int p = r;
  wp[(((size_t)co * (Cin >> 5) + cc) * 9 + tap) * 32 + p] = (_Float16)val;
}

// ---------------------------------------------------------------------------
// Conv3x3 (pad 1) + BN-bias + ReLU via WMMA f16->f32.
// Grid: (Cout/64, B, H).  Block: 256 threads = 8 waves.
// Wave wv: M-tile mi = wv&3, N-tiles ni0 = wv>>2 and ni0+2  (4x4 tile grid).
// LDS holds the 3 input rows, transposed to [row][wcol(66, halo)][ci(32)] so
// each lane's B fragment is one contiguous 32B load.  Halo columns and
// out-of-range rows are zero-filled once; interior is re-staged per K-chunk
// with vec4 global loads, software-pipelined one chunk ahead.
// ---------------------------------------------------------------------------
template <typename Tin>
__global__ __launch_bounds__(256) void conv3x3_bn_relu_wmma(
    const Tin* __restrict__ xin, const _Float16* __restrict__ wp,
    const float* __restrict__ bias, _Float16* __restrict__ out,
    int Cin, int Cout) {
  using T4 = typename Vec4<Tin>::type;
  __shared__ _Float16 ldsB[3][66][32];   // 12672 B

  const int tid  = threadIdx.x;
  const int lane = tid & 31;
  const int wv   = tid >> 5;
  const int l15  = lane & 15;
  const int lhalf = lane >> 4;           // 0: K 0..15 half, 1: K 16..31 half

  const int coutBase = blockIdx.x * 64;
  const int b = blockIdx.y;
  const int h = blockIdx.z;

  const int mi  = wv & 3;
  const int ni0 = wv >> 2;
  const int ni1 = ni0 + 2;
  const int chunks = Cin >> 5;

  // Zero-fill LDS once (covers halo columns + out-of-range rows forever).
  {
    unsigned long long* z = (unsigned long long*)&ldsB[0][0][0];
    for (int i = tid; i < 3 * 66 * 32 / 4; i += 256) z[i] = 0ull;
  }

  // Interior staging: 3 rows x 32 ci x 16 quads = 1536 quads = 6 per thread.
  T4 rv[6];
  auto stage_load = [&](int cc) {
#pragma unroll
    for (int s = 0; s < 6; ++s) {
      int idx = tid + s * 256;
      int dh  = idx >> 9;
      int rem = idx & 511;
      int ci  = rem >> 4;
      int w   = (rem & 15) << 2;
      int row = h + dh - 1;
      T4 v = {};
      if ((unsigned)row < 64u)
        v = *(const T4*)&xin[(((size_t)b * Cin + cc * 32 + ci) * 64 + row) * 64 + w];
      rv[s] = v;
    }
  };
  auto stage_store = [&]() {
#pragma unroll
    for (int s = 0; s < 6; ++s) {
      int idx = tid + s * 256;
      int dh  = idx >> 9;
      int rem = idx & 511;
      int ci  = rem >> 4;
      int w   = (rem & 15) << 2;
#pragma unroll
      for (int j = 0; j < 4; ++j)
        ldsB[dh][w + 1 + j][ci] = (_Float16)rv[s][j];
    }
  };

  v8f c0 = {}, c1 = {};
  stage_load(0);

  // Single A base pointer per wave; taps are immediate offsets (64 B apart),
  // chunks advance it by a constant 9*32 halves.
  const _Float16* aPtr = wp +
      ((size_t)(coutBase + mi * 16 + l15) * chunks) * 9 * 32 + lhalf * 16;

  for (int cc = 0; cc < chunks; ++cc) {
    __syncthreads();          // prior readers (or zero-fill) done
    stage_store();
    if (cc + 1 < chunks) stage_load(cc + 1);  // overlap next loads with WMMAs
    __syncthreads();          // staged data visible

#pragma unroll
    for (int tap = 0; tap < 9; ++tap) {
      const int dh = tap / 3;
      const int dw = tap % 3;
      // A fragment: contiguous 32B at immediate offset from the wave's base.
      v16h a = *(const v16h*)(aPtr + tap * 32);
      // B fragments from LDS: lane = column n, 16 contiguous k-halves.
      v16h bm0 = *(const v16h*)&ldsB[dh][ni0 * 16 + l15 + dw][lhalf * 16];
      v16h bm1 = *(const v16h*)&ldsB[dh][ni1 * 16 + l15 + dw][lhalf * 16];
      c0 = __builtin_amdgcn_wmma_f32_16x16x32_f16(false, a, false, bm0,
                                                  (short)0, c0, false, false);
      c1 = __builtin_amdgcn_wmma_f32_16x16x32_f16(false, a, false, bm1,
                                                  (short)0, c1, false, false);
    }
    aPtr += 9 * 32;
  }

  // Epilogue: bias + ReLU, store f16 NCHW.  C/D layout: elem (r,lane) ->
  // m = r + 8*(lane>=16), n = lane&15.
  float bi[8];
#pragma unroll
  for (int r = 0; r < 8; ++r)
    bi[r] = bias[coutBase + mi * 16 + r + 8 * lhalf];
#pragma unroll
  for (int r = 0; r < 8; ++r) {
    int co = coutBase + mi * 16 + r + 8 * lhalf;
    size_t baseo = (((size_t)b * Cout + co) * 64 + h) * 64;
    float v0 = c0[r] + bi[r]; v0 = v0 > 0.0f ? v0 : 0.0f;
    float v1 = c1[r] + bi[r]; v1 = v1 > 0.0f ? v1 : 0.0f;
    out[baseo + ni0 * 16 + l15] = (_Float16)v0;
    out[baseo + ni1 * 16 + l15] = (_Float16)v1;
  }
}

// ---------------------------------------------------------------------------
// 9x9 local-attention weighting: out[b,c,h,w] = sum_k atten[b,h,w,k] *
// x[b,c,h+kh-4,w+kw-4]  (zero-padded).  C = 64, 4 channels per thread.
// ---------------------------------------------------------------------------
__global__ __launch_bounds__(256) void weighting_kernel(
    const _Float16* __restrict__ x, const float* __restrict__ atten,
    _Float16* __restrict__ out) {
  int t = blockIdx.x * 256 + threadIdx.x;   // B*16*64*64 = 262144 total
  int w  = t & 63;
  int h  = (t >> 6) & 63;
  int cg = (t >> 12) & 15;
  int b  = t >> 16;
  float acc0 = 0.f, acc1 = 0.f, acc2 = 0.f, acc3 = 0.f;
  const float* ap = atten + (((size_t)(b * 64 + h) * 64 + w) * 81);
  const _Float16* xb = x + ((size_t)(b * 64 + cg * 4) << 12);
  for (int k = 0; k < 81; ++k) {
    int dh = k / 9 - 4, dw = k % 9 - 4;
    int hh = h + dh, ww = w + dw;
    if ((unsigned)hh >= 64u || (unsigned)ww >= 64u) continue;
    float a = ap[k];
    size_t o = (size_t)hh * 64 + ww;
    acc0 += a * (float)xb[o];
    acc1 += a * (float)xb[o + (1u << 12)];
    acc2 += a * (float)xb[o + (2u << 12)];
    acc3 += a * (float)xb[o + (3u << 12)];
  }
  size_t ob = ((size_t)(b * 64 + cg * 4) << 12) + (size_t)h * 64 + w;
  out[ob]              = (_Float16)acc0;
  out[ob + (1u << 12)] = (_Float16)acc1;
  out[ob + (2u << 12)] = (_Float16)acc2;
  out[ob + (3u << 12)] = (_Float16)acc3;
}

__global__ __launch_bounds__(256) void sub_kernel(
    const _Float16* __restrict__ a, const _Float16* __restrict__ b,
    _Float16* __restrict__ o, int n) {
  int i = blockIdx.x * 256 + threadIdx.x;
  if (i < n) o[i] = (_Float16)((float)a[i] - (float)b[i]);
}

// ---------------------------------------------------------------------------
// Global avg-pool + fc(256->10) + fc(10->1).  One block per batch element.
// ---------------------------------------------------------------------------
__global__ __launch_bounds__(256) void head_kernel(
    const _Float16* __restrict__ act, const float* __restrict__ fw1,
    const float* __restrict__ fb1, const float* __restrict__ fw2,
    const float* __restrict__ fb2, float* __restrict__ out) {
  int b = blockIdx.x;
  int c = threadIdx.x;
  const _Float16* p = act + ((size_t)(b * 256 + c) << 12);
  float s = 0.0f;
  for (int i = 0; i < 4096; ++i) s += (float)p[i];
  __shared__ float mean[256];
  __shared__ float o10[10];
  mean[c] = s * (1.0f / 4096.0f);
  __syncthreads();
  if (c < 10) {
    float acc = fb1[c];
    for (int k = 0; k < 256; ++k) acc += fw1[c * 256 + k] * mean[k];
    o10[c] = acc;
  }
  __syncthreads();
  if (c == 0) {
    float acc = fb2[0];
    for (int j = 0; j < 10; ++j) acc += fw2[j] * o10[j];
    out[b] = acc;
  }
}

// ---------------------------------------------------------------------------
extern "C" void kernel_launch(void* const* d_in, const int* in_sizes, int n_in,
                              void* d_out, int out_size, void* d_ws, size_t ws_size,
                              hipStream_t stream) {
  const float* low_key    = (const float*)d_in[0];
  const float* low_nonkey = (const float*)d_in[1];
  const float* atten      = (const float*)d_in[2];
  const float* w1  = (const float*)d_in[3];
  const float* bn1 = (const float*)d_in[4];
  const float* w2  = (const float*)d_in[5];
  const float* bn2 = (const float*)d_in[6];
  const float* w3  = (const float*)d_in[7];
  const float* bn3 = (const float*)d_in[8];
  const float* w4  = (const float*)d_in[9];
  const float* bn4 = (const float*)d_in[10];
  const float* w5  = (const float*)d_in[11];
  const float* bn5 = (const float*)d_in[12];
  const float* fw1 = (const float*)d_in[13];
  const float* fb1 = (const float*)d_in[14];
  const float* fw2 = (const float*)d_in[15];
  const float* fb2 = (const float*)d_in[16];

  size_t off = 0;
  auto carve = [&](size_t bytes) -> char* {
    size_t p = (off + 255) & ~(size_t)255;
    off = p + bytes;
    return (char*)d_ws + p;
  };

  const size_t HW = 64 * 64;
  _Float16* wp1 = (_Float16*)carve((size_t)9 * 256 * 1024 * 2);
  _Float16* wp2 = (_Float16*)carve((size_t)9 * 64 * 256 * 2);
  _Float16* wp3 = (_Float16*)carve((size_t)9 * 256 * 64 * 2);
  _Float16* wp4 = (_Float16*)carve((size_t)9 * 256 * 1024 * 2);
  _Float16* wp5 = (_Float16*)carve((size_t)9 * 256 * 256 * 2);
  float* bias1 = (float*)carve(256 * 4);
  float* bias2 = (float*)carve(64 * 4);
  float* bias3 = (float*)carve(256 * 4);
  float* bias4 = (float*)carve(256 * 4);
  float* bias5 = (float*)carve(256 * 4);
  _Float16* act1  = (_Float16*)carve((size_t)4 * 256 * HW * 2);
  _Float16* act2  = (_Float16*)carve((size_t)4 * 64 * HW * 2);
  _Float16* act2w = (_Float16*)carve((size_t)4 * 64 * HW * 2);
  _Float16* act3  = (_Float16*)carve((size_t)4 * 256 * HW * 2);
  _Float16* act4  = (_Float16*)carve((size_t)4 * 256 * HW * 2);
  _Float16* actd  = (_Float16*)carve((size_t)4 * 256 * HW * 2);
  _Float16* act5  = (_Float16*)carve((size_t)4 * 256 * HW * 2);
  (void)ws_size; (void)in_sizes; (void)n_in; (void)out_size;

  // --- weight packing (BN fold + f16 + [cout][cc][tap][32] repack) ---
  pack_weights_kernel<<<(9 * 256 * 1024 + 255) / 256, 256, 0, stream>>>(w1, bn1, wp1, bias1, 256, 1024);
  pack_weights_kernel<<<(9 * 64 * 256 + 255) / 256, 256, 0, stream>>>(w2, bn2, wp2, bias2, 64, 256);
  pack_weights_kernel<<<(9 * 256 * 64 + 255) / 256, 256, 0, stream>>>(w3, bn3, wp3, bias3, 256, 64);
  pack_weights_kernel<<<(9 * 256 * 1024 + 255) / 256, 256, 0, stream>>>(w4, bn4, wp4, bias4, 256, 1024);
  pack_weights_kernel<<<(9 * 256 * 256 + 255) / 256, 256, 0, stream>>>(w5, bn5, wp5, bias5, 256, 256);

  // --- pipeline ---
  conv3x3_bn_relu_wmma<float><<<dim3(4, 4, 64), 256, 0, stream>>>(low_key, wp1, bias1, act1, 1024, 256);
  conv3x3_bn_relu_wmma<_Float16><<<dim3(1, 4, 64), 256, 0, stream>>>(act1, wp2, bias2, act2, 256, 64);
  weighting_kernel<<<1024, 256, 0, stream>>>(act2, atten, act2w);
  conv3x3_bn_relu_wmma<_Float16><<<dim3(4, 4, 64), 256, 0, stream>>>(act2w, wp3, bias3, act3, 64, 256);
  conv3x3_bn_relu_wmma<float><<<dim3(4, 4, 64), 256, 0, stream>>>(low_nonkey, wp4, bias4, act4, 1024, 256);
  sub_kernel<<<(4 * 256 * (int)HW + 255) / 256, 256, 0, stream>>>(act3, act4, actd, 4 * 256 * (int)HW);
  conv3x3_bn_relu_wmma<_Float16><<<dim3(4, 4, 64), 256, 0, stream>>>(actd, wp5, bias5, act5, 256, 256);
  head_kernel<<<4, 256, 0, stream>>>(act5, fw1, fb1, fw2, fb2, (float*)d_out);
}
